// SpectralAttentionModule_14413910245914
// MI455X (gfx1250) — matrix operand order repaired and checked by reading, not compiled, not run on hardware
//
#include <hip/hip_runtime.h>

// ---------------- problem constants ----------------
#define B_    4
#define C_    64
#define D_    31
#define HW_   4096
#define NPB   (C_ * D_ * HW_)      // elements per batch = 8,126,464
#define NBLK  256                  // stats partial blocks per batch
#define EPS_  1e-5f

#define PTILE 4                    // spatial positions per workgroup
#define NCOL  128                  // padded columns (124 real = 31 d * 4 pos)
#define XSTR  72                   // LDS row stride (bf16) for x / attn-out / weights
#define QSTR  200                  // LDS col stride (bf16) for qkv (192 rows + pad)

// workspace byte offsets
#define WS_STATS_F   (2 * B_ * NBLK)       // float offset of mean/rstd pairs
#define WS_WQ_BYTES  16384                 // bf16 w_qkv (192*64 elems = 24,576 B)
#define WS_WP_BYTES  40960                 // bf16 w_proj (64*64 elems = 8,192 B)

typedef __attribute__((ext_vector_type(8)))  float  v8f;
typedef __attribute__((ext_vector_type(8)))  __bf16 v8bf;
typedef __attribute__((ext_vector_type(16))) __bf16 v16bf;
typedef int v4i __attribute__((vector_size(4 * sizeof(int))));

#define WMMA_BF16(A, B, C) \
  __builtin_amdgcn_wmma_f32_16x16x32_bf16(false, (A), false, (B), (short)0, (C), false, false)

#if __has_builtin(__builtin_amdgcn_global_load_async_to_lds_b128)
#define HAVE_ASYNC_LDS 1
// builtin expects (global int4*, local int4*, imm offset, imm cpol)
#define AS1V4(p) ((__attribute__((address_space(1))) v4i*)(p))
#define AS3V4(p) ((__attribute__((address_space(3))) v4i*)(p))
#endif

__device__ __forceinline__ void wait_async0() {
#if __has_builtin(__builtin_amdgcn_s_wait_asynccnt)
  __builtin_amdgcn_s_wait_asynccnt(0);
#else
  asm volatile("s_wait_asynccnt 0x0" ::: "memory");
#endif
}

__device__ __forceinline__ v16bf join16(v8bf lo, v8bf hi) {
  return __builtin_shufflevector(lo, hi, 0,1,2,3,4,5,6,7,8,9,10,11,12,13,14,15);
}
// A-matrix fragment (16x32 bf16): lane = row M (mod 16); K halves at
// kbase = kstep*32 + 8*(lane/16)  and  kbase+16   (ISA 7.12.2 16-bit A layout)
__device__ __forceinline__ v16bf load_afrag(const __bf16* row, int kbase) {
  v8bf lo = *(const v8bf*)(row + kbase);
  v8bf hi = *(const v8bf*)(row + kbase + 16);
  return join16(lo, hi);
}
// B-matrix fragment (32x16 bf16): lane = col N (mod 16); K halves at
// kbase = kstep*32 + 8*(lane/16)  and  kbase+16
__device__ __forceinline__ v16bf load_bfrag(const __bf16* colp, int kbase) {
  v8bf lo = *(const v8bf*)(colp + kbase);
  v8bf hi = *(const v8bf*)(colp + kbase + 16);
  return join16(lo, hi);
}

// ---------------- stage 1: per-batch partial sums (float4) ----------------
__launch_bounds__(256)
__global__ void sa_stats_partial(const float* __restrict__ x, float* __restrict__ ws) {
  const int blk = blockIdx.x;               // 0 .. B_*NBLK-1
  const int b   = blk / NBLK;
  const int r   = blk % NBLK;
  const int chunk = NPB / NBLK;             // 31,744
  const long base = (long)b * NPB + (long)r * chunk;
  const float4* xv = (const float4*)(x + base);
  const int chunk4 = chunk / 4;             // 7,936
  float s = 0.f, s2 = 0.f;
  for (int i = threadIdx.x; i < chunk4; i += 256) {
    float4 v = xv[i];
    s  += v.x + v.y + v.z + v.w;
    s2 += v.x * v.x + v.y * v.y + v.z * v.z + v.w * v.w;
  }
  __shared__ float red0[256], red1[256];
  red0[threadIdx.x] = s; red1[threadIdx.x] = s2;
  __syncthreads();
  for (int off = 128; off > 0; off >>= 1) {
    if (threadIdx.x < off) {
      red0[threadIdx.x] += red0[threadIdx.x + off];
      red1[threadIdx.x] += red1[threadIdx.x + off];
    }
    __syncthreads();
  }
  if (threadIdx.x == 0) {
    ws[2 * blk + 0] = red0[0];
    ws[2 * blk + 1] = red1[0];
  }
}

// ---------------- stage 1b: weights fp32 -> bf16 in workspace ----------------
__launch_bounds__(256)
__global__ void sa_wcvt(const float* __restrict__ w_qkv, const float* __restrict__ w_proj,
                        __bf16* __restrict__ wq16, __bf16* __restrict__ wp16) {
  const int idx = blockIdx.x * 256 + threadIdx.x;
  if (idx < 192 * 64) wq16[idx] = (__bf16)w_qkv[idx];
  const int j = idx - 192 * 64;
  if (j >= 0 && j < 64 * 64) wp16[j] = (__bf16)w_proj[j];
}

// ---------------- stage 2: finalize mean / rstd ----------------
__launch_bounds__(256)
__global__ void sa_stats_final(float* __restrict__ ws) {
  const int b = blockIdx.x;
  __shared__ float red0[256], red1[256];
  red0[threadIdx.x] = ws[2 * (b * NBLK + threadIdx.x) + 0];
  red1[threadIdx.x] = ws[2 * (b * NBLK + threadIdx.x) + 1];
  __syncthreads();
  for (int off = 128; off > 0; off >>= 1) {
    if (threadIdx.x < off) {
      red0[threadIdx.x] += red0[threadIdx.x + off];
      red1[threadIdx.x] += red1[threadIdx.x + off];
    }
    __syncthreads();
  }
  if (threadIdx.x == 0) {
    const float invn = 1.0f / (float)NPB;
    float mean = red0[0] * invn;
    float var  = red1[0] * invn - mean * mean;
    ws[WS_STATS_F + 2 * b + 0] = mean;
    ws[WS_STATS_F + 2 * b + 1] = rsqrtf(var + EPS_);
  }
}

// ---------------- stage 3: fused norm + QKV + attention + proj + residual ----------------
__launch_bounds__(256)
__global__ void sa_fused(const float* __restrict__ x,
                         const float* __restrict__ gamma,
                         const float* __restrict__ beta,
                         const float* __restrict__ w_qkv,
                         const float* __restrict__ b_qkv,
                         const float* __restrict__ w_proj,
                         const float* __restrict__ b_proj,
                         const float* __restrict__ stats,
                         const __bf16* __restrict__ wq16,
                         const __bf16* __restrict__ wp16,
                         float* __restrict__ out) {
  // LDS: ~123.5 KB total (fits 2 workgroups per 320KB WGP)
  __shared__ __attribute__((aligned(16))) __bf16 sWq[192 * XSTR];   // 27,648 B
  __shared__ __attribute__((aligned(16))) __bf16 sWp[64 * XSTR];    //  9,216 B
  __shared__ __attribute__((aligned(16))) __bf16 sX[NCOL * XSTR];   // 18,432 B  [col][ch]
  __shared__ __attribute__((aligned(16))) __bf16 sQKV[NCOL * QSTR]; // 51,200 B  [col][192ch]
  __shared__ __attribute__((aligned(16))) __bf16 sAO[NCOL * XSTR];  // 18,432 B  [col][ch]
  __shared__ float sBq[192], sBp[64], sG[64], sBe[64], sMS[2];

  const int tid  = threadIdx.x;
  const int b    = blockIdx.x >> 10;           // 1024 tiles per batch
  const int s0   = (blockIdx.x & 1023) * PTILE;
  const long xbase = (long)b * NPB;

  if (tid < 2) sMS[tid] = stats[2 * b + tid];
  if (tid < 64) { sG[tid] = gamma[tid]; sBe[tid] = beta[tid]; sBp[tid] = b_proj[tid]; }
  if (tid < 192) sBq[tid] = b_qkv[tid];

#ifdef HAVE_ASYNC_LDS
  // Async DMA of bf16 weights into padded LDS layout (ASYNCcnt path).
  // Overlaps with the VALU-heavy x-tile normalization below.
  for (int seg = tid; seg < 192 * 8; seg += 256) {       // 16B segments, 6 iters
    const int o = seg >> 3, s8 = seg & 7;
    __builtin_amdgcn_global_load_async_to_lds_b128(
        AS1V4(wq16 + seg * 8), AS3V4(&sWq[o * XSTR + s8 * 8]), 0, 0);
  }
  for (int seg = tid; seg < 64 * 8; seg += 256) {        // 2 iters
    const int o = seg >> 3, s8 = seg & 7;
    __builtin_amdgcn_global_load_async_to_lds_b128(
        AS1V4(wp16 + seg * 8), AS3V4(&sWp[o * XSTR + s8 * 8]), 0, 0);
  }
#else
  for (int i = tid; i < 192 * 64; i += 256) {
    int o = i >> 6, c = i & 63;
    sWq[o * XSTR + c] = (__bf16)w_qkv[i];
  }
  for (int i = tid; i < 64 * 64; i += 256) {
    int o = i >> 6, c = i & 63;
    sWp[o * XSTR + c] = (__bf16)w_proj[i];
  }
#endif
  __syncthreads();                 // guards sMS / sG / sBe for the normalize loop
  const float mean = sMS[0], rstd = sMS[1];

  // zero pad columns 124..127
  { int col = 124 + (tid >> 6), c = tid & 63; sX[col * XSTR + c] = (__bf16)0.f; }
  // x tile -> normalize -> LDS bf16, [col][channel]; 4 contiguous spatial
  // positions per (c,d) pair => one float4 global load each.
  for (int i = tid; i < C_ * D_; i += 256) {   // 1,984 pairs, ~8 iters
    const int c = i / D_;
    const int d = i - c * D_;
    const float4 x4 = *(const float4*)(&x[xbase + ((long)c * D_ + d) * HW_ + s0]);
    const float vx[4] = {x4.x, x4.y, x4.z, x4.w};
    const float g = sG[c] * rstd, be = sBe[c] - mean * g;
#pragma unroll
    for (int p = 0; p < 4; ++p) {
      sX[(d * 4 + p) * XSTR + c] = (__bf16)(vx[p] * g + be);
    }
  }
#ifdef HAVE_ASYNC_LDS
  wait_async0();                   // weight DMA must land before GEMM reads LDS
#endif
  __syncthreads();

  const int wave  = tid >> 5;
  const int lane  = tid & 31;
  const int khalf = lane >> 4;                 // 0/1
  const int lcol  = wave * 16 + (lane & 15);   // this wave's N-tile column

  // ---- QKV GEMM: qkv[192 x 128] = Wq[192x64] @ Xn[64x128], K = 2 x 32 ----
  {
    const __bf16* pb = &sX[lcol * XSTR];
    v16bf bf0 = load_bfrag(pb, 0  + 8 * khalf);
    v16bf bf1 = load_bfrag(pb, 32 + 8 * khalf);
    for (int mt = 0; mt < 12; ++mt) {
      const __bf16* pa = &sWq[(mt * 16 + (lane & 15)) * XSTR];
      v16bf af0 = load_afrag(pa, 0  + 8 * khalf);
      v16bf af1 = load_afrag(pa, 32 + 8 * khalf);
      v8f acc = {};
      acc = WMMA_BF16(af0, bf0, acc);
      acc = WMMA_BF16(af1, bf1, acc);
      // D layout: VGPR r -> row  mt*16 + 8*khalf + r,  col lcol
      const int o0 = mt * 16 + 8 * khalf;
      v8bf pk;
#pragma unroll
      for (int r = 0; r < 8; ++r) pk[r] = (__bf16)(acc[r] + sBq[o0 + r]);
      *(v8bf*)(&sQKV[lcol * QSTR + o0]) = pk;   // packed ds_store_b128
    }
  }
  __syncthreads();

  // ---- attention: wave = head, lane = query depth d, loop 4 positions ----
  {
    const int head = wave;
    const int d    = lane;
    const float scale = 0.35355339059327373f;  // 1/sqrt(8)
    if (d < D_) {
      for (int p = 0; p < PTILE; ++p) {
        v8bf qv = *(const v8bf*)(&sQKV[(d * PTILE + p) * QSTR + head * 8]);
        float q[8];
#pragma unroll
        for (int h = 0; h < 8; ++h) q[h] = (float)qv[h];
        float s[D_];
        float mx = -3.0e38f;
#pragma unroll
        for (int j = 0; j < D_; ++j) {
          v8bf kv = *(const v8bf*)(&sQKV[(j * PTILE + p) * QSTR + 64 + head * 8]);
          float dot = 0.f;
#pragma unroll
          for (int h = 0; h < 8; ++h) dot += q[h] * (float)kv[h];
          dot *= scale;
          s[j] = dot;
          mx = fmaxf(mx, dot);
        }
        float sum = 0.f;
#pragma unroll
        for (int j = 0; j < D_; ++j) { s[j] = __expf(s[j] - mx); sum += s[j]; }
        const float inv = 1.0f / sum;
        float o[8] = {0.f, 0.f, 0.f, 0.f, 0.f, 0.f, 0.f, 0.f};
#pragma unroll
        for (int j = 0; j < D_; ++j) {
          v8bf vv = *(const v8bf*)(&sQKV[(j * PTILE + p) * QSTR + 128 + head * 8]);
#pragma unroll
          for (int h = 0; h < 8; ++h) o[h] += s[j] * (float)vv[h];
        }
        v8bf pk;
#pragma unroll
        for (int h = 0; h < 8; ++h) pk[h] = (__bf16)(o[h] * inv);
        *(v8bf*)(&sAO[(d * PTILE + p) * XSTR + head * 8]) = pk;
      }
    }
  }
  __syncthreads();

  // ---- proj GEMM + bias + residual + store: y[64 x 128] = Wp[64x64] @ AO[64x128] ----
  {
    const __bf16* pb = &sAO[lcol * XSTR];
    v16bf bf0 = load_bfrag(pb, 0  + 8 * khalf);
    v16bf bf1 = load_bfrag(pb, 32 + 8 * khalf);
    const int dcol = lcol >> 2, pcol = lcol & 3;
    for (int mt = 0; mt < 4; ++mt) {
      const __bf16* pa = &sWp[(mt * 16 + (lane & 15)) * XSTR];
      v16bf af0 = load_afrag(pa, 0  + 8 * khalf);
      v16bf af1 = load_afrag(pa, 32 + 8 * khalf);
      v8f acc = {};
      acc = WMMA_BF16(af0, bf0, acc);
      acc = WMMA_BF16(af1, bf1, acc);
      if (lcol < D_ * PTILE) {
        const int o0 = mt * 16 + 8 * khalf;
#pragma unroll
        for (int r = 0; r < 8; ++r) {
          const int o = o0 + r;
          const long gi = xbase + ((long)o * D_ + dcol) * HW_ + s0 + pcol;
          out[gi] = x[gi] + acc[r] + sBp[o];
        }
      }
    }
  }
}

// ---------------- launcher ----------------
extern "C" void kernel_launch(void* const* d_in, const int* in_sizes, int n_in,
                              void* d_out, int out_size, void* d_ws, size_t ws_size,
                              hipStream_t stream) {
  const float* x      = (const float*)d_in[0];
  const float* gamma  = (const float*)d_in[1];
  const float* beta   = (const float*)d_in[2];
  const float* w_qkv  = (const float*)d_in[3];
  const float* b_qkv  = (const float*)d_in[4];
  const float* w_proj = (const float*)d_in[5];
  const float* b_proj = (const float*)d_in[6];
  float* out = (float*)d_out;
  float* ws  = (float*)d_ws;
  __bf16* wq16 = (__bf16*)((char*)d_ws + WS_WQ_BYTES);
  __bf16* wp16 = (__bf16*)((char*)d_ws + WS_WP_BYTES);

  sa_stats_partial<<<B_ * NBLK, 256, 0, stream>>>(x, ws);
  sa_wcvt<<<64, 256, 0, stream>>>(w_qkv, w_proj, wq16, wp16);
  sa_stats_final<<<B_, 256, 0, stream>>>(ws);
  const float* stats = ws + WS_STATS_F;
  sa_fused<<<B_ * (HW_ / PTILE), 256, 0, stream>>>(x, gamma, beta, w_qkv, b_qkv,
                                                   w_proj, b_proj, stats,
                                                   wq16, wp16, out);
}
